// QLSTM_65481071395833
// MI455X (gfx1250) — matrix operand order repaired and checked by reading, not compiled, not run on hardware
//
#include <hip/hip_runtime.h>

#define SEQ   512
#define BATCH 64
#define DIN   1024
#define DH    1024
#define K2    2048
// packed weights: gates(4) * jtiles(64) * ktiles(64) * lanes(32) * elems(16)
#define WP_ELEMS   (4u * 64u * 64u * 32u * 16u)          // 8388608
// packed x fragments per step: mtiles(4) * ktiles(32) * lanes(32) * elems(16)
#define XF_STEP    (4u * 32u * 32u * 16u)                // 65536
#define XF_ELEMS   ((unsigned)SEQ * XF_STEP)             // 33554432

typedef __attribute__((ext_vector_type(16))) __bf16          bf16x16;
typedef __attribute__((ext_vector_type(16))) unsigned short  u16x16;
typedef __attribute__((ext_vector_type(8)))  float           f32x8;

static __device__ __forceinline__ unsigned short f2bf(float f) {
  unsigned int u = __float_as_uint(f);
  u += 0x7fffu + ((u >> 16) & 1u);
  return (unsigned short)(u >> 16);
}

// ---------------------------------------------------------------------------
// Pack fp32 weights (row-major [n,k], k spans [x;h]) into WMMA B-fragment
// order [gate][jt][kt][lane][e]:  n = jt*16 + (lane&15),
//                                 k = kt*32 + (lane>=16 ? 8 : 0) + e + (e>=8 ? 8 : 0)
// ---------------------------------------------------------------------------
__global__ __launch_bounds__(256) void pack_weights(
    const float* __restrict__ Wf, const float* __restrict__ Wi,
    const float* __restrict__ Wg, const float* __restrict__ Wo,
    unsigned short* __restrict__ wp) {
  unsigned int idx  = blockIdx.x * 256u + threadIdx.x;
  unsigned int e    = idx & 15u;
  unsigned int lane = (idx >> 4) & 31u;
  unsigned int kt   = (idx >> 9) & 63u;
  unsigned int jt   = (idx >> 15) & 63u;
  unsigned int g    = idx >> 21;
  const float* W = (g == 0u) ? Wf : (g == 1u) ? Wi : (g == 2u) ? Wg : Wo;
  unsigned int n = jt * 16u + (lane & 15u);
  unsigned int k = kt * 32u + ((lane >> 4) << 3) + e + ((e >= 8u) ? 8u : 0u);
  wp[idx] = f2bf(W[n * (unsigned)K2 + k]);
}

// ---------------------------------------------------------------------------
// Pack all x_t into WMMA A-fragment order [t][mtile][kt][lane][e]:
//   m = mtile*16 + (lane&15)
//   k_local = hi*8 + e + (e>=8 ? 8 : 0)   (hi = lane>=16)
// ---------------------------------------------------------------------------
__global__ __launch_bounds__(256) void pack_x(const float* __restrict__ x,
                                              unsigned short* __restrict__ xf) {
  unsigned int idx   = blockIdx.x * 256u + threadIdx.x;
  unsigned int e     = idx & 15u;
  unsigned int lane  = (idx >> 4) & 31u;
  unsigned int kt    = (idx >> 9) & 31u;
  unsigned int mtile = (idx >> 14) & 3u;
  unsigned int t     = idx >> 16;
  unsigned int hi    = lane >> 4;
  unsigned int m     = mtile * 16u + (lane & 15u);
  unsigned int kl    = hi * 8u + e + ((e >= 8u) ? 8u : 0u);
  xf[idx] = f2bf(x[(size_t)t * BATCH * DIN + (size_t)m * DIN + kt * 32u + kl]);
}

__global__ __launch_bounds__(256) void zero_c(float* __restrict__ c) {
  c[blockIdx.x * 256u + threadIdx.x] = 0.0f;
}

__global__ __launch_bounds__(256) void finalize(const float* __restrict__ out,
                                                const float* __restrict__ c,
                                                float* __restrict__ hn,
                                                float* __restrict__ cn) {
  unsigned int i = blockIdx.x * 256u + threadIdx.x;
  hn[i] = out[(size_t)(SEQ - 1) * BATCH * DH + i];
  cn[i] = c[i];
}

// ---------------------------------------------------------------------------
// One LSTM timestep. Block = 128 threads = 4 waves; wave g computes gate g of
// one 16x16 (batch x hidden) tile. Inner loop is pure 32B-load + WMMA.
// Epilogue fuses sigmoid/tanh/c-update, writes h (fp32) and the bf16
// A-fragment-ordered h for the next step.
// ---------------------------------------------------------------------------
__global__ __launch_bounds__(128) void lstm_step(
    const unsigned short* __restrict__ xf_t,   // packed x frags for this t
    const unsigned short* __restrict__ hf_in,  // packed h frags (prev step)
    unsigned short* __restrict__ hf_out,       // packed h frags (this step)
    int has_h,
    const unsigned short* __restrict__ wp,
    const float* __restrict__ bF, const float* __restrict__ bI,
    const float* __restrict__ bG, const float* __restrict__ bO,
    float* __restrict__ c,                     // (BATCH, DH) fp32, in/out
    float* __restrict__ h_out)                 // (BATCH, DH) fp32
{
  __shared__ float zt[4][16][17];

  const int lane  = threadIdx.x & 31;
  const int g     = threadIdx.x >> 5;
  const int mtile = blockIdx.x >> 6;
  const int jt    = blockIdx.x & 63;
  const int hi    = lane >> 4;
  const int nloc  = lane & 15;

  const float* bias_p = (g == 0) ? bF : (g == 1) ? bI : (g == 2) ? bG : bO;
  const float  bias   = bias_p[jt * 16 + nloc];

  f32x8 acc;
#pragma unroll
  for (int r = 0; r < 8; ++r) acc[r] = bias;

  // fragment base pointers (32B-aligned, contiguous per lane)
  const unsigned short* ax = xf_t + (size_t)mtile * (32u * 512u) + lane * 16;
  const unsigned short* ah = hf_in + (size_t)mtile * (32u * 512u) + lane * 16;
  const unsigned short* wb = wp + ((size_t)(g * 64 + jt) * 64) * 512 + lane * 16;

  // x contribution: K = 0..1023
#pragma unroll 8
  for (int kt = 0; kt < 32; ++kt) {
    bf16x16 a = *(const bf16x16*)(ax + (size_t)kt * 512);
    bf16x16 b = *(const bf16x16*)(wb + (size_t)kt * 512);
    acc = __builtin_amdgcn_wmma_f32_16x16x32_bf16(false, a, false, b,
                                                  (short)0, acc, false, false);
  }
  // h contribution: K = 1024..2047 (h0 == 0 -> skipped at t=0; wave-uniform)
  if (has_h) {
#pragma unroll 8
    for (int kt = 0; kt < 32; ++kt) {
      bf16x16 a = *(const bf16x16*)(ah + (size_t)kt * 512);
      bf16x16 b = *(const bf16x16*)(wb + (size_t)(32 + kt) * 512);
      acc = __builtin_amdgcn_wmma_f32_16x16x32_bf16(false, a, false, b,
                                                    (short)0, acc, false, false);
    }
  }

  // gate tile -> LDS (C/D layout: VGPR r, lane halves)
#pragma unroll
  for (int r = 0; r < 8; ++r)
    zt[g][r + hi * 8][nloc] = acc[r];
  __syncthreads();

  // fused nonlinearity + state update, plus bf16 h-fragment scatter
#pragma unroll
  for (int p = threadIdx.x; p < 256; p += 128) {
    const int mr = p >> 4, nc = p & 15;
    const int gm = mtile * 16 + mr;          // batch index
    const int gn = jt * 16 + nc;             // hidden index
    const float fz = zt[0][mr][nc];
    const float iz = zt[1][mr][nc];
    const float gz = zt[2][mr][nc];
    const float oz = zt[3][mr][nc];
    const float fs = 1.0f / (1.0f + __expf(-fz));
    const float is = 1.0f / (1.0f + __expf(-iz));
    const float gt = tanhf(gz);
    const float os = 1.0f / (1.0f + __expf(-oz));
    const size_t ci = (size_t)gm * DH + gn;
    const float cn = fs * c[ci] + is * gt;
    c[ci] = cn;
    const float hv = os * tanhf(cn);
    h_out[ci] = hv;
    // scatter h into A-fragment order for the next step:
    //   mtile' = gm>>4, row = gm&15, kt' = gn>>5, kl = gn&31
    //   hi' = (kl>>3)&1, e' = (kl&7) + ((kl>>4)<<3)
    const int kl  = gn & 31;
    const int hi2 = (kl >> 3) & 1;
    const int e2  = (kl & 7) + ((kl >> 4) << 3);
    const size_t fidx =
        (((size_t)(gm >> 4) * 32 + (gn >> 5)) * 32 + hi2 * 16 + (gm & 15)) * 16 + e2;
    hf_out[fidx] = f2bf(hv);
  }
}

extern "C" void kernel_launch(void* const* d_in, const int* in_sizes, int n_in,
                              void* d_out, int out_size, void* d_ws, size_t ws_size,
                              hipStream_t stream) {
  const float* x   = (const float*)d_in[0];
  const float* Wf  = (const float*)d_in[1];
  const float* bF  = (const float*)d_in[2];
  const float* Wi  = (const float*)d_in[3];
  const float* bI  = (const float*)d_in[4];
  const float* Wg  = (const float*)d_in[5];
  const float* bG  = (const float*)d_in[6];
  const float* Wo  = (const float*)d_in[7];
  const float* bO  = (const float*)d_in[8];
  float* out = (float*)d_out;

  char* ws = (char*)d_ws;
  unsigned short* wp = (unsigned short*)ws;                       // 16 MB
  unsigned short* xf = (unsigned short*)(ws + (size_t)WP_ELEMS * 2);  // 64 MB
  unsigned short* hf0 = (unsigned short*)(ws + (size_t)WP_ELEMS * 2
                                             + (size_t)XF_ELEMS * 2);   // 128 KB
  unsigned short* hf1 = hf0 + XF_STEP;                            // 128 KB
  float* c = (float*)(hf1 + XF_STEP);                             // 256 KB

  pack_weights<<<WP_ELEMS / 256, 256, 0, stream>>>(Wf, Wi, Wg, Wo, wp);
  pack_x<<<XF_ELEMS / 256, 256, 0, stream>>>(x, xf);
  zero_c<<<(BATCH * DH) / 256, 256, 0, stream>>>(c);

  for (int t = 0; t < SEQ; ++t) {
    const unsigned short* xf_t  = xf + (size_t)t * XF_STEP;
    const unsigned short* hf_in = (t & 1) ? hf1 : hf0;   // read prev
    unsigned short*       hf_ot = (t & 1) ? hf0 : hf1;   // write this
    lstm_step<<<256, 128, 0, stream>>>(xf_t, hf_in, hf_ot, (t > 0) ? 1 : 0, wp,
                                       bF, bI, bG, bO,
                                       c, out + (size_t)t * BATCH * DH);
  }

  float* hn = out + (size_t)SEQ * BATCH * DH;
  float* cn = hn + (size_t)BATCH * DH;
  finalize<<<(BATCH * DH) / 256, 256, 0, stream>>>(out, c, hn, cn);
}
